// VSSBlock_23948737643093
// MI455X (gfx1250) — matrix operand order repaired and checked by reading, not compiled, not run on hardware
//
#include <hip/hip_runtime.h>

// ---------------- CDNA5 WMMA types ----------------
typedef __attribute__((ext_vector_type(16))) __bf16 v16bf;
typedef __attribute__((ext_vector_type(8)))  float  v8f;

__device__ __forceinline__ __bf16 f2bf(float f) {
    unsigned u = __builtin_bit_cast(unsigned, f);
    unsigned r = u + 0x7FFFu + ((u >> 16) & 1u);      // round-to-nearest-even
    unsigned short h = (unsigned short)(r >> 16);
    return __builtin_bit_cast(__bf16, h);
}
__device__ __forceinline__ float bf2f(__bf16 h) {
    unsigned short u = __builtin_bit_cast(unsigned short, h);
    return __builtin_bit_cast(float, ((unsigned)u) << 16);
}
__device__ __forceinline__ float silu(float x) { return x / (1.f + __expf(-x)); }

// ---------------- f32 -> bf16 conversion ----------------
__global__ __launch_bounds__(256)
void cvt_f32_bf16(const float* __restrict__ in, __bf16* __restrict__ out, int n) {
    int i = blockIdx.x * 256 + threadIdx.x;
    if (i < n) out[i] = f2bf(in[i]);
}

// ---------------- LayerNorm (D=384), bf16 out ----------------
__global__ __launch_bounds__(384)
void ln_kernel(const float* __restrict__ x, const float* __restrict__ w,
               const float* __restrict__ b, __bf16* __restrict__ out) {
    const int D = 384;
    int m = blockIdx.x, t = threadIdx.x;
    float v = x[(size_t)m * D + t];
    float s = v, s2 = v * v;
    #pragma unroll
    for (int off = 16; off; off >>= 1) {
        s  += __shfl_xor(s,  off, 32);
        s2 += __shfl_xor(s2, off, 32);
    }
    __shared__ float sh[2][12];
    int wv = t >> 5, lane = t & 31;
    if (lane == 0) { sh[0][wv] = s; sh[1][wv] = s2; }
    __syncthreads();
    if (t == 0) {
        float S = 0.f, S2 = 0.f;
        #pragma unroll
        for (int i = 0; i < 12; i++) { S += sh[0][i]; S2 += sh[1][i]; }
        sh[0][0] = S / (float)D;
        sh[1][0] = S2 / (float)D;
    }
    __syncthreads();
    float mu  = sh[0][0];
    float var = sh[1][0] - mu * mu;
    float xn = (v - mu) * rsqrtf(var + 1e-6f) * w[t] + b[t];
    out[(size_t)m * D + t] = f2bf(xn);
}

// ---------------- bf16 WMMA GEMM: C[M,N] = A[M,K] * W[N,K]^T ----------------
// Double-buffered LDS, A tile via gfx1250 async global->LDS DMA (ASYNCcnt),
// pipelined so tile(it+1) DMA overlaps WMMA on tile(it).
// B tile staged via registers (needs N-bounds zero-fill).
// mode 0: store f32   mode 1: store bf16   mode 2: store f32 = resid + C
__global__ __launch_bounds__(256)
void gemm_bf16_wmma(const __bf16* __restrict__ A, const __bf16* __restrict__ W,
                    float* __restrict__ Cf, __bf16* __restrict__ Cbf,
                    const float* __restrict__ resid,
                    int M, int N, int K, int mode) {
    constexpr int BM = 128, BN = 64, BK = 32, LDA = 40, LDB = 40;
    __shared__ __bf16 As[2][BM * LDA];   // 2 x 10 KB
    __shared__ __bf16 Bs[2][BN * LDB];   // 2 x  5 KB

    const int m0 = blockIdx.x * BM, n0 = blockIdx.y * BN;
    const int tid = threadIdx.x, lane = tid & 31, wv = tid >> 5;
    const int wm = (wv >> 1) * 32, wn = (wv & 1) * 32;  // 4x2 wave grid of 32x32 tiles
    const int lrow = lane & 15, hi = lane >> 4;

    const int arow = tid >> 1, aseg = (tid & 1) * 16;
    const int brow = tid >> 2, bcg  = (tid & 3) * 8;
    const bool bok = (n0 + brow) < N;

    const __bf16* gA = A + (size_t)(m0 + arow) * K + aseg;
    const __bf16* gW = W + (size_t)(n0 + brow) * K + bcg;

    // LDS byte address for async DMA dest: low 32 bits of generic pointer
    // are the LDS aperture offset (ISA 10.2).
    const unsigned ldsA[2] = {
        (unsigned)(uintptr_t)&As[0][arow * LDA + aseg],
        (unsigned)(uintptr_t)&As[1][arow * LDA + aseg]
    };

    const int T = K / BK;

    auto issueA = [&](int t) {  // 32 B/thread of A tile t -> As[t&1]  [ASYNCcnt +2]
        const __bf16* p = gA + t * BK;
        asm volatile("global_load_async_to_lds_b128 %0, %1, off\n\t"
                     "global_load_async_to_lds_b128 %0, %1, off offset:16"
                     :: "v"(ldsA[t & 1]), "v"(p) : "memory");
    };
    auto loadB = [&](int t) -> uint4 {
        uint4 v = {0u, 0u, 0u, 0u};
        if (bok) v = *(const uint4*)(gW + t * BK);
        return v;
    };

    union Frag { uint4 u[2]; v16bf v; };
    v8f acc[2][2] = {};

    // ---- pipeline prologue: tile0 resident, tile1 in flight ----
    issueA(0);
    uint4 bv = loadB(0);
    *(uint4*)&Bs[0][brow * LDB + bcg] = bv;
    if (T > 1) {
        issueA(1);
        bv = loadB(1);
        asm volatile("s_wait_asynccnt 0x2" ::: "memory");  // tile0 landed
    } else {
        asm volatile("s_wait_asynccnt 0x0" ::: "memory");
    }
    __syncthreads();

    for (int it = 0; it < T; ++it) {
        const int buf = it & 1;
        const __bf16* as = &As[buf][0];
        const __bf16* bs = &Bs[buf][0];

        // A fragment layout (16-bit 16x32, ISA 7.12.2):
        //   lanes 0-15 : K = {0..7, 16..23},  lanes 16-31: K = {8..15, 24..31}
        Frag a[2], b[2];
        const int k0 = hi * 8;
        #pragma unroll
        for (int i = 0; i < 2; i++) {
            int r = wm + i * 16 + lrow;
            a[i].u[0] = *(const uint4*)&as[r * LDA + k0];
            a[i].u[1] = *(const uint4*)&as[r * LDA + k0 + 16];
        }
        // B fragment: lanes 0-15 hold K={0..7,16..23}, lanes 16-31 hold K={8..15,24..31}
        const int kb = hi * 8;
        #pragma unroll
        for (int j = 0; j < 2; j++) {
            int c = wn + j * 16 + lrow;
            b[j].u[0] = *(const uint4*)&bs[c * LDB + kb];
            b[j].u[1] = *(const uint4*)&bs[c * LDB + kb + 16];
        }
        #pragma unroll
        for (int i = 0; i < 2; i++)
            #pragma unroll
            for (int j = 0; j < 2; j++)
                acc[i][j] = __builtin_amdgcn_wmma_f32_16x16x32_bf16(
                    false, a[i].v, false, b[j].v, (short)0, acc[i][j], false, false);

        __syncthreads();  // all waves done reading buf

        if (it + 1 < T) {
            // stage tile(it+1) B into the buffer just freed for writing
            *(uint4*)&Bs[buf ^ 1][brow * LDB + bcg] = bv;
            if (it + 2 < T) {
                issueA(it + 2);          // DMA overlaps next compute
                bv = loadB(it + 2);
                if (it + 3 < T)          // prefetch the tile after that
                    __builtin_prefetch((const void*)(gA + (it + 3) * BK), 0, 3);
                asm volatile("s_wait_asynccnt 0x2" ::: "memory");  // tile(it+1) landed
            } else {
                asm volatile("s_wait_asynccnt 0x0" ::: "memory");
            }
            __syncthreads();
        }
    }

    // C/D layout: VGPR r -> M = r + 8*hi, N = lane&15
    #pragma unroll
    for (int i = 0; i < 2; i++) {
        #pragma unroll
        for (int j = 0; j < 2; j++) {
            int mg = m0 + wm + i * 16 + hi * 8;
            int ng = n0 + wn + j * 16 + lrow;
            if (ng < N) {
                #pragma unroll
                for (int r = 0; r < 8; r++) {
                    float v = acc[i][j][r];
                    size_t o = (size_t)(mg + r) * N + ng;
                    if (mode == 0)      Cf[o]  = v;
                    else if (mode == 1) Cbf[o] = f2bf(v);
                    else                Cf[o]  = resid[o] + v;
                }
            }
        }
    }
}

// ---------------- depthwise causal conv (K=4) + SiLU ----------------
__global__ __launch_bounds__(256)
void conv_silu_kernel(const __bf16* __restrict__ xz, const float* __restrict__ cw,
                      const float* __restrict__ cb, __bf16* __restrict__ xc,
                      int Bn, int L, int E) {
    long idx = (long)blockIdx.x * 256 + threadIdx.x;
    long total = (long)Bn * L * E;
    if (idx >= total) return;
    int e = (int)(idx % E);
    long ml = idx / E;
    int l = (int)(ml % L);
    long brow = ml - l;  // b*L
    float acc = cb[e];
    #pragma unroll
    for (int k = 0; k < 4; k++) {
        int lp = l + k - 3;
        if (lp >= 0) acc += cw[e * 4 + k] * bf2f(xz[(brow + lp) * 2 * E + e]);
    }
    xc[idx] = f2bf(silu(acc));
}

// ---------------- dt = softplus(dbc[:,:24] @ dtw^T + dtb) ----------------
__global__ __launch_bounds__(256)
void dt_kernel(const float* __restrict__ dbc, const float* __restrict__ dtw,
               const float* __restrict__ dtb, float* __restrict__ dt, int M) {
    __shared__ float rs[24];
    int m = blockIdx.x;
    if (threadIdx.x < 24) rs[threadIdx.x] = dbc[(size_t)m * 152 + threadIdx.x];
    __syncthreads();
    for (int e = threadIdx.x; e < 768; e += 256) {
        float acc = dtb[e];
        #pragma unroll
        for (int r = 0; r < 24; r++) acc += rs[r] * dtw[e * 24 + r];
        float sp = (acc > 20.f) ? acc : log1pf(__expf(acc));
        dt[(size_t)m * 768 + e] = sp;
    }
}

// ---------------- selective scan + skip + SiLU(z) gate ----------------
// one wave32 per (b,e); 2 state elements per lane (N=64)
__global__ __launch_bounds__(256)
void scan_kernel(const float* __restrict__ dt, const __bf16* __restrict__ xc,
                 const float* __restrict__ dbc, const __bf16* __restrict__ xz,
                 const float* __restrict__ A_log, const float* __restrict__ Dsk,
                 __bf16* __restrict__ ym, int Bn, int L, int E) {
    int wv = threadIdx.x >> 5, lane = threadIdx.x & 31;
    int w = blockIdx.x * 8 + wv;             // 0..B*E-1
    int b = w / E, e = w % E;
    float A0 = -__expf(A_log[e * 64 + lane]);
    float A1 = -__expf(A_log[e * 64 + lane + 32]);
    float Dv = Dsk[e];
    float h0 = 0.f, h1 = 0.f;
    size_t base = (size_t)b * L;
    for (int t = 0; t < L; t++) {
        size_t row = base + t;
        float dtv = dt[row * 768 + e];
        float xv  = bf2f(xc[row * 768 + e]);
        const float* bc = dbc + row * 152;
        float Bv0 = bc[24  + lane], Bv1 = bc[56  + lane];
        float Cv0 = bc[88  + lane], Cv1 = bc[120 + lane];
        h0 = h0 * __expf(dtv * A0) + dtv * xv * Bv0;
        h1 = h1 * __expf(dtv * A1) + dtv * xv * Bv1;
        float s = h0 * Cv0 + h1 * Cv1;
        #pragma unroll
        for (int off = 16; off; off >>= 1) s += __shfl_xor(s, off, 32);
        if (lane == 0) {
            float y = s + Dv * xv;
            float zv = bf2f(xz[row * 1536 + 768 + e]);
            y *= silu(zv);
            ym[row * 768 + e] = f2bf(y);
        }
    }
}

// ---------------- launcher ----------------
extern "C" void kernel_launch(void* const* d_in, const int* in_sizes, int n_in,
                              void* d_out, int out_size, void* d_ws, size_t ws_size,
                              hipStream_t stream) {
    const float* input      = (const float*)d_in[0];
    const float* ln_w       = (const float*)d_in[1];
    const float* ln_b       = (const float*)d_in[2];
    const float* in_proj_w  = (const float*)d_in[3];
    const float* conv_w     = (const float*)d_in[4];
    const float* conv_b     = (const float*)d_in[5];
    const float* x_proj_w   = (const float*)d_in[6];
    const float* dt_proj_w  = (const float*)d_in[7];
    const float* dt_proj_b  = (const float*)d_in[8];
    const float* A_log      = (const float*)d_in[9];
    const float* D_skip     = (const float*)d_in[10];
    const float* out_proj_w = (const float*)d_in[11];

    const int Bn = 8, L = 2048, D = 384, E = 768, R2N = 152;
    const int M = Bn * L;  // 16384

    char* ws = (char*)d_ws;
    size_t off = 0;
    auto alloc = [&](size_t bytes) -> void* {
        off = (off + 255) & ~(size_t)255;
        void* p = ws + off;
        off += bytes;
        return p;
    };
    __bf16* xn_bf = (__bf16*)alloc((size_t)M * D * 2);
    __bf16* w_in  = (__bf16*)alloc((size_t)2 * E * D * 2);
    __bf16* w_xp  = (__bf16*)alloc((size_t)R2N * E * 2);
    __bf16* w_out = (__bf16*)alloc((size_t)D * E * 2);
    __bf16* xz_bf = (__bf16*)alloc((size_t)M * 2 * E * 2);
    __bf16* xc_bf = (__bf16*)alloc((size_t)M * E * 2);
    float*  dbc   = (float*) alloc((size_t)M * R2N * 4);
    float*  dtb   = (float*) alloc((size_t)M * E * 4);
    __bf16* ym_bf = (__bf16*)alloc((size_t)M * E * 2);

    // weights -> bf16
    int n1 = 2 * E * D, n2 = R2N * E, n3 = D * E;
    cvt_f32_bf16<<<(n1 + 255) / 256, 256, 0, stream>>>(in_proj_w,  w_in,  n1);
    cvt_f32_bf16<<<(n2 + 255) / 256, 256, 0, stream>>>(x_proj_w,   w_xp,  n2);
    cvt_f32_bf16<<<(n3 + 255) / 256, 256, 0, stream>>>(out_proj_w, w_out, n3);

    // LayerNorm
    ln_kernel<<<M, 384, 0, stream>>>(input, ln_w, ln_b, xn_bf);

    // in_proj: (M,384) x (1536,384)^T -> xz bf16 (M,1536)
    gemm_bf16_wmma<<<dim3(M / 128, (2 * E + 63) / 64), 256, 0, stream>>>(
        xn_bf, w_in, nullptr, xz_bf, nullptr, M, 2 * E, D, 1);

    // depthwise causal conv + SiLU
    long tot = (long)M * E;
    conv_silu_kernel<<<(int)((tot + 255) / 256), 256, 0, stream>>>(
        xz_bf, conv_w, conv_b, xc_bf, Bn, L, E);

    // x_proj: (M,768) x (152,768)^T -> dbc f32 (M,152)
    gemm_bf16_wmma<<<dim3(M / 128, (R2N + 63) / 64), 256, 0, stream>>>(
        xc_bf, w_xp, dbc, nullptr, nullptr, M, R2N, E, 0);

    // dt projection + softplus (exact f32, K=24)
    dt_kernel<<<M, 256, 0, stream>>>(dbc, dt_proj_w, dt_proj_b, dtb, M);

    // selective scan + D-skip + SiLU(z) gate -> ym bf16
    scan_kernel<<<(Bn * E) / 8, 256, 0, stream>>>(
        dtb, xc_bf, dbc, xz_bf, A_log, D_skip, ym_bf, Bn, L, E);

    // out_proj + residual: (M,768) x (384,768)^T + input -> d_out f32
    gemm_bf16_wmma<<<dim3(M / 128, D / 64), 256, 0, stream>>>(
        ym_bf, w_out, (float*)d_out, nullptr, input, M, D, E, 2);
}
